// UpSampler8Stage_16518444220823
// MI455X (gfx1250) — matrix operand order repaired and checked by reading, not compiled, not run on hardware
//
#include <hip/hip_runtime.h>

#define NLOW 16384
#define CCH  32
#define MTOT (8 * NLOW)

typedef __attribute__((ext_vector_type(16))) __bf16 v16bf;
typedef __attribute__((ext_vector_type(8)))  float  v8f;

struct __attribute__((aligned(16))) U4 { unsigned int x, y, z, w; };

union V16U { U4 q[2]; v16bf v; };

static __device__ __forceinline__ v16bf mkv(U4 a, U4 b) {
    V16U u; u.q[0] = a; u.q[1] = b; return u.v;
}

// f32 -> bf16 round-to-nearest-even
static __device__ __forceinline__ unsigned short f2bf(float f) {
    unsigned int u = __float_as_uint(f);
    u += 0x7FFFu + ((u >> 16) & 1u);
    return (unsigned short)(u >> 16);
}

// ---------------------------------------------------------------------------
// utility kernels
// ---------------------------------------------------------------------------
__global__ void zero_f32(float* p, int n) {
    int i = blockIdx.x * blockDim.x + threadIdx.x;
    if (i < n) p[i] = 0.0f;
}
__global__ void fill_f32(float* p, float v, int n) {
    int i = blockIdx.x * blockDim.x + threadIdx.x;
    if (i < n) p[i] = v;
}
__global__ void zero_u16(unsigned short* p, int n) {
    int i = blockIdx.x * blockDim.x + threadIdx.x;
    if (i < n) p[i] = 0;
}

// Convert f32 taps [T,32,32] into WMMA-B bf16 layout [T][nh=2][lane=32][e=16].
// B element (K=cin, N=cout): lane half 0 holds K = e (0..15), half 1 holds K = 16+e.
__global__ void prep_w(const float* __restrict__ W, unsigned short* __restrict__ out, int ntap) {
    int tid = blockIdx.x * blockDim.x + threadIdx.x;
    if (tid >= ntap * 1024) return;
    int t    = tid >> 10;
    int r    = tid & 1023;
    int nh   = r >> 9;
    int lane = (r >> 4) & 31;
    int e    = r & 15;
    int half = lane >> 4;
    int n    = nh * 16 + (lane & 15);
    int kk   = half * 16 + e;
    out[((t * 2 + nh) * 32 + lane) * 16 + e] = f2bf(W[t * 1024 + kk * 32 + n]);
}

// block_in first conv: Cin=1 -> C=32, all-valid, ReLU.  w0 layout [27,1,32].
__global__ void conv0_low(const float* __restrict__ x, const int* __restrict__ nbr,
                          const float* __restrict__ w0, const float* __restrict__ b0,
                          float* __restrict__ outf, unsigned short* __restrict__ outb) {
    int n = blockIdx.x * blockDim.x + threadIdx.x;
    if (n >= NLOW) return;
    float acc[CCH];
    for (int c = 0; c < CCH; ++c) acc[c] = b0[c];
    for (int k = 0; k < 27; ++k) {
        int idx = nbr[n * 27 + k];
        if (idx >= 0) {
            float v = x[idx];
            for (int c = 0; c < CCH; ++c) acc[c] += v * w0[k * CCH + c];
        }
    }
    for (int c = 0; c < CCH; ++c) {
        float o = fmaxf(acc[c], 0.0f);
        outf[n * CCH + c] = o;
        outb[n * CCH + c] = f2bf(o);
    }
}

// ---------------------------------------------------------------------------
// Core sparse conv: gather-GEMM-scatter on v_wmma_f32_16x16x32_bf16.
// One wave = 32 output rows x 32 channels; 27 taps x (2 A-tiles x 2 B-tiles)
// = 108 WMMAs per wave.  Weights staged in LDS in B layout; A gathered in the
// contiguous k order k = 16*half + e, as in B (two b128 loads/lane).  Absent neighbors remap to
// a dedicated zero row at index Mrows (single cndmask, unconditional loads).
// ---------------------------------------------------------------------------
__global__ void __launch_bounds__(256) conv_wmma(
    const unsigned short* __restrict__ fin,   // [Mrows+16,32] bf16 (row Mrows = 0)
    const unsigned short* __restrict__ wB,    // [27][2][32][16] bf16 (prepped)
    const float* __restrict__ bias,           // [32]
    const float* __restrict__ mask,           // [Mrows] 1.0 / 0.0
    const int*   __restrict__ nbr,            // [Mrows,27]
    const float* __restrict__ addsrc,         // [Mrows,32] or nullptr (residual)
    float*          __restrict__ outf,        // [Mrows,32] or nullptr
    unsigned short* __restrict__ outb,        // [Mrows,32] or nullptr
    int relu, int Mrows)
{
    __shared__ U4 sW[27 * 2 * 32 * 2];        // 3456 x 16B = 54 KB
    const U4* gW = (const U4*)wB;
    for (int i = threadIdx.x; i < 3456; i += 256) sW[i] = gW[i];
    __syncthreads();

    const int lane = threadIdx.x & 31;
    const int wv   = threadIdx.x >> 5;
    const int half = lane >> 4;
    const int nlo  = lane & 15;
    const int base = (blockIdx.x * 8 + wv) * 32;  // 32 rows per wave
    const U4* fq = (const U4*)fin;

    v8f acc[2][2] = {{{}, {}}, {{}, {}}};

    const int g0 = (base + nlo) * 27;
    const int g1 = (base + 16 + nlo) * 27;

    for (int k = 0; k < 27; ++k) {
        int i0 = nbr[g0 + k];
        int i1 = nbr[g1 + k];
        i0 = (i0 < 0) ? Mrows : i0;           // zero-row remap
        i1 = (i1 < 0) ? Mrows : i1;
        const U4* rp0 = fq + (i0 << 2);
        const U4* rp1 = fq + (i1 << 2);
        v16bf A0 = mkv(rp0[2 * half], rp0[2 * half + 1]);
        v16bf A1 = mkv(rp1[2 * half], rp1[2 * half + 1]);
        int w0i = ((k * 2 + 0) * 32 + lane) * 2;
        int w1i = ((k * 2 + 1) * 32 + lane) * 2;
        v16bf B0 = mkv(sW[w0i], sW[w0i + 1]);
        v16bf B1 = mkv(sW[w1i], sW[w1i + 1]);
        acc[0][0] = __builtin_amdgcn_wmma_f32_16x16x32_bf16(false, A0, false, B0,
                                                            (short)0, acc[0][0], false, false);
        acc[0][1] = __builtin_amdgcn_wmma_f32_16x16x32_bf16(false, A0, false, B1,
                                                            (short)0, acc[0][1], false, false);
        acc[1][0] = __builtin_amdgcn_wmma_f32_16x16x32_bf16(false, A1, false, B0,
                                                            (short)0, acc[1][0], false, false);
        acc[1][1] = __builtin_amdgcn_wmma_f32_16x16x32_bf16(false, A1, false, B1,
                                                            (short)0, acc[1][1], false, false);
    }

    const float b0 = bias[nlo];
    const float b1 = bias[nlo + 16];
#pragma unroll
    for (int t = 0; t < 2; ++t) {
#pragma unroll
        for (int r = 0; r < 8; ++r) {
            int row = base + t * 16 + half * 8 + r;  // D layout: VGPR r -> M=r (+8 hi half)
            float vm = mask[row];
            float o0 = vm * (acc[t][0][r] + b0);
            float o1 = vm * (acc[t][1][r] + b1);
            if (relu) { o0 = fmaxf(o0, 0.0f); o1 = fmaxf(o1, 0.0f); }
            if (addsrc) { o0 += addsrc[row * 32 + nlo]; o1 += addsrc[row * 32 + nlo + 16]; }
            if (outf) {
                outf[row * 32 + nlo]      = o0;
                outf[row * 32 + nlo + 16] = o1;
            }
            if (outb) {
                outb[row * 32 + nlo]      = f2bf(o0);
                outb[row * 32 + nlo + 16] = f2bf(o1);
            }
        }
    }
}

// generative transpose conv k2 s2: feats_u[i*8+o] = h[i] @ wt[o] + bt
__global__ void __launch_bounds__(256) tconv(const float* __restrict__ h,
                                             const float* __restrict__ wt,
                                             const float* __restrict__ bt,
                                             float* __restrict__ outF) {
    __shared__ float swt[8 * 32 * 32];
    __shared__ float sbt[32];
    for (int i = threadIdx.x; i < 8192; i += 256) swt[i] = wt[i];
    if (threadIdx.x < 32) sbt[threadIdx.x] = bt[threadIdx.x];
    __syncthreads();
    int m = blockIdx.x * 256 + threadIdx.x;
    int i = m >> 3, o = m & 7;
    float f[CCH];
    for (int c = 0; c < CCH; ++c) f[c] = h[i * CCH + c];
    for (int d = 0; d < CCH; ++d) {
        float a = sbt[d];
        for (int c = 0; c < CCH; ++c) a += f[c] * swt[(o * 32 + c) * 32 + d];
        outF[m * CCH + d] = a;
    }
}

// stage prep: in_feat = out_mask*hs_prev + (octant==so)*feats_u ; in_mask = or()
__global__ void prep_stage(const int* __restrict__ octant, int so,
                           const float* __restrict__ out_mask,
                           const float* __restrict__ hs,
                           const float* __restrict__ feats,
                           unsigned short* __restrict__ in_bf,
                           float* __restrict__ in_mask) {
    int m = blockIdx.x * 256 + threadIdx.x;
    int sl = (octant[m] == so) ? 1 : 0;
    float om = out_mask[m];
    in_mask[m] = fmaxf(om, sl ? 1.0f : 0.0f);
    for (int c = 0; c < CCH; ++c) {
        float v = om * hs[m * CCH + c] + (sl ? feats[m * CCH + c] : 0.0f);
        in_bf[m * CCH + c] = f2bf(v);
    }
}

// classifier on sl rows + out_mask update (in_mask & gt)
__global__ void __launch_bounds__(256) post_cls(
    const float* __restrict__ hs, const float* __restrict__ in_mask,
    const int* __restrict__ octant, int so, const unsigned char* __restrict__ gt,
    const float* __restrict__ cw0, const float* __restrict__ cb0,
    const float* __restrict__ cw1, const float* __restrict__ cb1,
    const float* __restrict__ cw2, const float* __restrict__ cb2,
    float* __restrict__ outp, float* __restrict__ out_mask)
{
    __shared__ float s0[1024], s1[1024], s2[32], sb0[32], sb1[32];
    for (int i = threadIdx.x; i < 1024; i += 256) { s0[i] = cw0[i]; s1[i] = cw1[i]; }
    if (threadIdx.x < 32) {
        s2[threadIdx.x]  = cw2[threadIdx.x];
        sb0[threadIdx.x] = cb0[threadIdx.x];
        sb1[threadIdx.x] = cb1[threadIdx.x];
    }
    __syncthreads();
    int m = blockIdx.x * 256 + threadIdx.x;
    out_mask[m] = in_mask[m] * (gt[m] ? 1.0f : 0.0f);
    float res = 0.0f;
    if (octant[m] == so) {
        float h1[CCH];
        for (int j = 0; j < CCH; ++j) {
            float a = sb0[j];
            for (int c = 0; c < CCH; ++c) a += hs[m * CCH + c] * s0[c * 32 + j];
            h1[j] = fmaxf(a, 0.0f);
        }
        for (int j = 0; j < CCH; ++j) {
            float a = sb1[j];
            for (int c = 0; c < CCH; ++c) a += h1[c] * s1[c * 32 + j];
            res += fmaxf(a, 0.0f) * s2[j];
        }
        res += cb2[0];
    }
    outp[m] = res;
}

// ---------------------------------------------------------------------------
extern "C" void kernel_launch(void* const* d_in, const int* in_sizes, int n_in,
                              void* d_out, int out_size, void* d_ws, size_t ws_size,
                              hipStream_t stream) {
    const float* x_low  = (const float*)d_in[0];
    const float* bi_w0  = (const float*)d_in[1];
    const float* bi_b0  = (const float*)d_in[2];
    const float* bi_w   = (const float*)d_in[3];
    const float* bi_b   = (const float*)d_in[4];
    const float* wt     = (const float*)d_in[5];
    const float* bt     = (const float*)d_in[6];
    const float* sw     = (const float*)d_in[7];
    const float* sb     = (const float*)d_in[8];
    const float* cw0    = (const float*)d_in[9];
    const float* cb0    = (const float*)d_in[10];
    const float* cw1    = (const float*)d_in[11];
    const float* cb1    = (const float*)d_in[12];
    const float* cw2    = (const float*)d_in[13];
    const float* cb2    = (const float*)d_in[14];
    const int* nbr_low  = (const int*)d_in[15];
    const int* nbr_u    = (const int*)d_in[16];
    const int* octant   = (const int*)d_in[17];
    const unsigned char* gt = (const unsigned char*)d_in[18];
    float* outp = (float*)d_out;

    const int M = MTOT;
    const int TAPB = 27 * 1024; // bf16 elements per conv layer (27 taps x 2 x 32 x 16)

    char* ws = (char*)d_ws;
    size_t off = 0;
    auto carve = [&](size_t bytes) -> char* {
        char* p = ws + off;
        off = (off + bytes + 255) & ~(size_t)255;
        return p;
    };

    // bf16 feature buffers get +16 rows; row [Mrows] is the zero row for -1 taps
    unsigned short* bi_wp = (unsigned short*)carve((size_t)7 * TAPB * 2);
    unsigned short* sw_p  = (unsigned short*)carve((size_t)64 * TAPB * 2);
    float* ones_low       = (float*)carve((size_t)NLOW * 4);
    float* hloA_f         = (float*)carve((size_t)NLOW * 32 * 4);
    unsigned short* hloA_b= (unsigned short*)carve((size_t)(NLOW + 16) * 32 * 2);
    float* hloB_f         = (float*)carve((size_t)NLOW * 32 * 4);
    unsigned short* hloB_b= (unsigned short*)carve((size_t)(NLOW + 16) * 32 * 2);
    float* tlo_f          = (float*)carve((size_t)NLOW * 32 * 4);
    unsigned short* tlo_b = (unsigned short*)carve((size_t)(NLOW + 16) * 32 * 2);
    float* feats          = (float*)carve((size_t)M * 32 * 4);
    unsigned short* in_bf = (unsigned short*)carve((size_t)(M + 16) * 32 * 2);
    float* in_mask        = (float*)carve((size_t)M * 4);
    float* out_mask       = (float*)carve((size_t)M * 4);
    float* hA_f           = (float*)carve((size_t)M * 32 * 4);
    unsigned short* hA_b  = (unsigned short*)carve((size_t)(M + 16) * 32 * 2);
    float* hB_f           = (float*)carve((size_t)M * 32 * 4);
    unsigned short* hB_b  = (unsigned short*)carve((size_t)(M + 16) * 32 * 2);
    unsigned short* t_b   = (unsigned short*)carve((size_t)(M + 16) * 32 * 2);
    float* hs_f           = (float*)carve((size_t)M * 32 * 4);

    // --- init state read at stage 0 + zero rows for -1 remap ---
    zero_f32<<<(M * 32 + 255) / 256, 256, 0, stream>>>(hs_f, M * 32);
    zero_f32<<<(M + 255) / 256, 256, 0, stream>>>(out_mask, M);
    fill_f32<<<(NLOW + 255) / 256, 256, 0, stream>>>(ones_low, 1.0f, NLOW);
    zero_u16<<<1, 32, 0, stream>>>(hloA_b + (size_t)NLOW * 32, 32);
    zero_u16<<<1, 32, 0, stream>>>(hloB_b + (size_t)NLOW * 32, 32);
    zero_u16<<<1, 32, 0, stream>>>(tlo_b + (size_t)NLOW * 32, 32);
    zero_u16<<<1, 32, 0, stream>>>(in_bf + (size_t)M * 32, 32);
    zero_u16<<<1, 32, 0, stream>>>(hA_b + (size_t)M * 32, 32);
    zero_u16<<<1, 32, 0, stream>>>(hB_b + (size_t)M * 32, 32);
    zero_u16<<<1, 32, 0, stream>>>(t_b + (size_t)M * 32, 32);

    // --- weight prep into WMMA-B bf16 layout ---
    prep_w<<<(7 * 27 * 1024 + 255) / 256, 256, 0, stream>>>(bi_w, bi_wp, 7 * 27);
    prep_w<<<(64 * 27 * 1024 + 255) / 256, 256, 0, stream>>>(sw, sw_p, 64 * 27);

    // --- block_in at tensor_stride 2 ---
    conv0_low<<<(NLOW + 255) / 256, 256, 0, stream>>>(x_low, nbr_low, bi_w0, bi_b0,
                                                      hloA_f, hloA_b);
    {
        float* hf = hloA_f; unsigned short* hb = hloA_b;
        float* gf = hloB_f; unsigned short* gb = hloB_b;
        for (int i = 0; i < 3; ++i) {
            conv_wmma<<<NLOW / 256, 256, 0, stream>>>(
                hb, bi_wp + (2 * i) * TAPB, bi_b + (2 * i) * 32, ones_low, nbr_low,
                nullptr, nullptr, tlo_b, 1, NLOW);
            conv_wmma<<<NLOW / 256, 256, 0, stream>>>(
                tlo_b, bi_wp + (2 * i + 1) * TAPB, bi_b + (2 * i + 1) * 32, ones_low,
                nbr_low, hf, gf, gb, 0, NLOW);
            { float* tf = hf; hf = gf; gf = tf; }
            { unsigned short* tb2 = hb; hb = gb; gb = tb2; }
        }
        conv_wmma<<<NLOW / 256, 256, 0, stream>>>(
            hb, bi_wp + 6 * TAPB, bi_b + 6 * 32, ones_low, nbr_low,
            nullptr, tlo_f, nullptr, 0, NLOW);
    }

    // --- generative transpose conv ---
    tconv<<<M / 256, 256, 0, stream>>>(tlo_f, wt, bt, feats);

    // --- 8 stages ---
    const int order[8] = {0, 7, 1, 6, 5, 2, 3, 4};
    for (int s = 0; s < 8; ++s) {
        prep_stage<<<M / 256, 256, 0, stream>>>(octant, order[s], out_mask, hs_f,
                                                feats, in_bf, in_mask);
        const unsigned short* swb = sw_p + (size_t)(s * 8) * TAPB;
        const float* sbb = sb + (size_t)(s * 8) * 32;

        float* hf = hA_f; unsigned short* hb = hA_b;
        float* gf = hB_f; unsigned short* gb = hB_b;
        conv_wmma<<<M / 256, 256, 0, stream>>>(in_bf, swb, sbb, in_mask, nbr_u,
                                               nullptr, hf, hb, 1, M);
        for (int i = 0; i < 3; ++i) {
            conv_wmma<<<M / 256, 256, 0, stream>>>(
                hb, swb + (1 + 2 * i) * TAPB, sbb + (1 + 2 * i) * 32, in_mask, nbr_u,
                nullptr, nullptr, t_b, 1, M);
            conv_wmma<<<M / 256, 256, 0, stream>>>(
                t_b, swb + (2 + 2 * i) * TAPB, sbb + (2 + 2 * i) * 32, in_mask, nbr_u,
                hf, gf, gb, 0, M);
            { float* tf = hf; hf = gf; gf = tf; }
            { unsigned short* tb2 = hb; hb = gb; gb = tb2; }
        }
        conv_wmma<<<M / 256, 256, 0, stream>>>(hb, swb + 7 * TAPB, sbb + 7 * 32,
                                               in_mask, nbr_u, nullptr, hs_f, nullptr,
                                               0, M);
        post_cls<<<M / 256, 256, 0, stream>>>(hs_f, in_mask, octant, order[s], gt,
                                              cw0, cb0, cw1, cb1, cw2, cb2,
                                              outp + (size_t)s * M, out_mask);
    }
    (void)in_sizes; (void)n_in; (void)out_size; (void)ws_size;
}
